// StackDecoderLayer_25701084299506
// MI455X (gfx1250) — matrix operand order repaired and checked by reading, not compiled, not run on hardware
//
#include <hip/hip_runtime.h>
#include <hip/hip_bf16.h>
#include <stdint.h>

// ---------------------------------------------------------------------------
// StackDecoderLayer forward for MI455X (gfx1250, wave32, WMMA bf16)
// ---------------------------------------------------------------------------

typedef unsigned short u16;
typedef __attribute__((ext_vector_type(16))) __bf16      bf16x16;
typedef __attribute__((ext_vector_type(8)))  float       f32x8;
typedef __attribute__((ext_vector_type(4)))  unsigned int v4u;
typedef int v4i __attribute__((vector_size(16)));   // matches async-LDS builtin param

#define AS_GLOBAL __attribute__((address_space(1)))
#define AS_LDS    __attribute__((address_space(3)))

union Frag { v4u q[2]; bf16x16 v; };

#define SEQL   512
#define BATCH  16
#define EMB    512
#define NTOK   8192      /* SEQL*BATCH */
#define NHEAD  8
#define HD     64
#define DHID   512
#define DFF    2048
#define DSS    128
#define SDEP   48
#define SWID   128

// CDNA5 async global->LDS staging path (guarded: falls back to reg staging)
#if defined(__gfx1250__) && __has_builtin(__builtin_amdgcn_global_load_async_to_lds_b128) && __has_builtin(__builtin_amdgcn_s_wait_asynccnt)
#define USE_ASYNC_LDS 1
#else
#define USE_ASYNC_LDS 0
#endif

// ---------------- helpers ----------------

static __device__ __forceinline__ u16 f2bf(float f) {
  unsigned u = __float_as_uint(f);
  unsigned r = u + 0x7FFFu + ((u >> 16) & 1u);
  return (u16)(r >> 16);
}
static __device__ __forceinline__ float bf2f(u16 v) {
  return __uint_as_float(((unsigned)v) << 16);
}

static __device__ __forceinline__ float warp_sum(float v) {
#pragma unroll
  for (int m = 16; m >= 1; m >>= 1) v += __shfl_xor(v, m, 32);
  return v;
}
static __device__ __forceinline__ float warp_max(float v) {
#pragma unroll
  for (int m = 16; m >= 1; m >>= 1) v = fmaxf(v, __shfl_xor(v, m, 32));
  return v;
}
static __device__ __forceinline__ float half_sum(float v) {   // within 16-lane half
#pragma unroll
  for (int m = 8; m >= 1; m >>= 1) v += __shfl_xor(v, m, 32);
  return v;
}
static __device__ __forceinline__ float half_max(float v) {
#pragma unroll
  for (int m = 8; m >= 1; m >>= 1) v = fmaxf(v, __shfl_xor(v, m, 32));
  return v;
}

static __device__ __forceinline__ f32x8 wmma_bf16(const bf16x16& a, const bf16x16& b, const f32x8& c) {
  return __builtin_amdgcn_wmma_f32_16x16x32_bf16(false, a, false, b, (short)0, c, false, false);
}

// ---------------- elementwise kernels ----------------

__global__ void k_cvt_bf16(const float* __restrict__ in, u16* __restrict__ out, int n) {
  int i = blockIdx.x * blockDim.x + threadIdx.x;
  if (i < n) out[i] = f2bf(in[i]);
}

// layer norm over rows of 512; one wave per row; bf16 output
__global__ __launch_bounds__(256) void k_ln(const float* __restrict__ x,
                                            const float* __restrict__ g,
                                            const float* __restrict__ bb,
                                            u16* __restrict__ out, int rows) {
  const int wid = threadIdx.x >> 5, lane = threadIdx.x & 31;
  const int row = blockIdx.x * 8 + wid;
  if (row >= rows) return;
  const float* xr = x + (size_t)row * EMB;
  float4 v[4];
  float s = 0.f, s2 = 0.f;
#pragma unroll
  for (int i = 0; i < 4; i++) {
    v[i] = *(const float4*)(xr + i * 128 + lane * 4);
    s  += v[i].x + v[i].y + v[i].z + v[i].w;
    s2 += v[i].x*v[i].x + v[i].y*v[i].y + v[i].z*v[i].z + v[i].w*v[i].w;
  }
  s = warp_sum(s); s2 = warp_sum(s2);
  const float mean = s * (1.f / EMB);
  const float var  = s2 * (1.f / EMB) - mean * mean;
  const float rstd = rsqrtf(var + 1e-5f);
#pragma unroll
  for (int i = 0; i < 4; i++) {
    const int c = i * 128 + lane * 4;
    ushort4 o;
    o.x = f2bf((v[i].x - mean) * rstd * g[c+0] + bb[c+0]);
    o.y = f2bf((v[i].y - mean) * rstd * g[c+1] + bb[c+1]);
    o.z = f2bf((v[i].z - mean) * rstd * g[c+2] + bb[c+2]);
    o.w = f2bf((v[i].w - mean) * rstd * g[c+3] + bb[c+3]);
    *(ushort4*)(out + (size_t)row * EMB + c) = o;
  }
}

// split qkv (t-order 8192x1536 bf16) into Qh/Kh [B,H,S,hd] and VhT [B,H,hd,S] bf16
__global__ void k_qkv_split(const u16* __restrict__ qkv, u16* __restrict__ Qh,
                            u16* __restrict__ Kh, u16* __restrict__ VhT) {
  int i = blockIdx.x * blockDim.x + threadIdx.x;
  if (i >= NTOK * EMB) return;
  const int t = i >> 9, c = i & 511;
  const int b = t & 15, s = t >> 4;
  const int h = c >> 6, d = c & 63;
  const size_t base = (size_t)t * (3 * EMB);
  const float q = bf2f(qkv[base + c]) * 0.125f;       // hd^-0.5
  const u16 k = qkv[base + EMB + c];
  const u16 v = qkv[base + 2 * EMB + c];
  const size_t bh = (size_t)(b * NHEAD + h);
  Qh[(bh * SEQL + s) * HD + d]  = f2bf(q);
  Kh[(bh * SEQL + s) * HD + d]  = k;
  VhT[(bh * HD + d) * SEQL + s] = v;
}

// gather x1 (t-order f32) into u-order bf16
__global__ void k_gather_u(const float* __restrict__ x1, u16* __restrict__ out) {
  int i = blockIdx.x * blockDim.x + threadIdx.x;
  if (i >= NTOK * EMB) return;
  const int u = i >> 9, c = i & 511;
  const int b = u >> 9, s = u & 511;
  const int t = s * BATCH + b;
  out[i] = f2bf(x1[(size_t)t * EMB + c]);
}

// stack_prev[:,:,0,:] -> bf16 (u-order, 8192x128)
__global__ void k_stackx(const float* __restrict__ sp, u16* __restrict__ out) {
  int i = blockIdx.x * blockDim.x + threadIdx.x;
  if (i >= NTOK * SWID) return;
  const int u = i >> 7, c = i & 127;
  out[i] = f2bf(sp[(size_t)u * SDEP * SWID + c]);
}

// Newton iteration for inverse of y^3/3 + y (== nonsat)
__global__ void k_nonsat(const float* __restrict__ in, float* __restrict__ out,
                         u16* __restrict__ obf, int n) {
  int i = blockIdx.x * blockDim.x + threadIdx.x;
  if (i >= n) return;
  const float xs = in[i];
  float y = xs;
#pragma unroll
  for (int it = 0; it < 12; ++it) {
    const float y2 = y * y;
    y = (0.6666667f * y2 * y + xs) / (y2 + 1.f);
  }
  out[i] = y;
  if (obf) obf[i] = f2bf(y);
}

// vx = concat([x1 (t-order), hidden (u-order)], -1) as bf16, rows t-order x 1024
__global__ void k_vx_build(const float* __restrict__ x1, const float* __restrict__ hid,
                           u16* __restrict__ vx) {
  int i = blockIdx.x * blockDim.x + threadIdx.x;
  if (i >= NTOK * (EMB + DHID)) return;
  const int t = i >> 10, c = i & 1023;
  float v;
  if (c < EMB) v = x1[(size_t)t * EMB + c];
  else {
    const int b = t & 15, s = t >> 4;
    const int u = b * SEQL + s;
    v = hid[(size_t)u * DHID + (c - EMB)];
  }
  vx[i] = f2bf(v);
}

// softmax over rows of 128, one wave per row, bf16 out
__global__ __launch_bounds__(256) void k_softmax128(const float* __restrict__ z,
                                                    u16* __restrict__ p, int rows) {
  const int wid = threadIdx.x >> 5, lane = threadIdx.x & 31;
  const int row = blockIdx.x * 8 + wid;
  if (row >= rows) return;
  const float4 v = *(const float4*)(z + (size_t)row * DSS + lane * 4);
  float m = fmaxf(fmaxf(v.x, v.y), fmaxf(v.z, v.w));
  m = warp_max(m);
  const float ex = __expf(v.x - m), ey = __expf(v.y - m);
  const float ez = __expf(v.z - m), ew = __expf(v.w - m);
  float s = warp_sum(ex + ey + ez + ew);
  const float inv = 1.f / s;
  ushort4 o;
  o.x = f2bf(ex * inv); o.y = f2bf(ey * inv); o.z = f2bf(ez * inv); o.w = f2bf(ew * inv);
  *(ushort4*)(p + (size_t)row * DSS + lane * 4) = o;
}

// controls = softmax(hidden @ A_w.T + A_b) ; one wave per row (u-order)
__global__ __launch_bounds__(256) void k_controls(const float* __restrict__ hid,
                                                  const float* __restrict__ Aw,
                                                  const float* __restrict__ Ab,
                                                  float* __restrict__ ctrl, int rows) {
  const int wid = threadIdx.x >> 5, lane = threadIdx.x & 31;
  const int row = blockIdx.x * 8 + wid;
  if (row >= rows) return;
  const float* hr = hid + (size_t)row * DHID;
  float c0 = 0.f, c1 = 0.f, c2 = 0.f;
  for (int k = lane; k < DHID; k += 32) {
    const float h = hr[k];
    c0 += h * Aw[k];
    c1 += h * Aw[DHID + k];
    c2 += h * Aw[2 * DHID + k];
  }
  c0 = warp_sum(c0) + Ab[0];
  c1 = warp_sum(c1) + Ab[1];
  c2 = warp_sum(c2) + Ab[2];
  const float m = fmaxf(c0, fmaxf(c1, c2));
  const float e0 = __expf(c0 - m), e1 = __expf(c1 - m), e2 = __expf(c2 - m);
  const float inv = 1.f / (e0 + e1 + e2);
  if (lane == 0) {
    ctrl[row * 3 + 0] = e0 * inv;
    ctrl[row * 3 + 1] = e1 * inv;
    ctrl[row * 3 + 2] = e2 * inv;
  }
}

// stack = a_noop*prev + a_push*[inp;prev[:-1]] + a_pop*[prev[1:];0]
__global__ void k_stack_update(const float* __restrict__ sp, const float* __restrict__ sinp,
                               const float* __restrict__ ctrl, float* __restrict__ out) {
  int i = blockIdx.x * blockDim.x + threadIdx.x;
  const int total = NTOK * SDEP * (SWID / 4);
  if (i >= total) return;
  const int u = i / (SDEP * 32);
  const int r = i - u * SDEP * 32;
  const int d = r >> 5, c4 = r & 31;
  const float a_push = ctrl[u * 3 + 0];
  const float a_pop  = ctrl[u * 3 + 1];
  const float a_noop = ctrl[u * 3 + 2];
  const size_t base = ((size_t)u * SDEP + d) * SWID + c4 * 4;
  const float4 pv = *(const float4*)(sp + base);
  float4 up, dn;
  if (d == 0) up = *(const float4*)(sinp + (size_t)u * SWID + c4 * 4);
  else        up = *(const float4*)(sp + base - SWID);
  if (d == SDEP - 1) dn = make_float4(0.f, 0.f, 0.f, 0.f);
  else               dn = *(const float4*)(sp + base + SWID);
  float4 o;
  o.x = a_noop * pv.x + a_push * up.x + a_pop * dn.x;
  o.y = a_noop * pv.y + a_push * up.y + a_pop * dn.y;
  o.z = a_noop * pv.z + a_push * up.z + a_pop * dn.z;
  o.w = a_noop * pv.w + a_push * up.w + a_pop * dn.w;
  *(float4*)(out + base) = o;
}

// ---------------- WMMA GEMM: C = A[M,K](bf16) @ W[N,K]^T (bf16) + epilogue ----
// flags: 1 = relu(acc+bias), 2 = accumulate into existing C, 4 = C = 0.5*(acc+res)
#define GF_RELU 1
#define GF_ACC  2
#define GF_HALF 4

__global__ __launch_bounds__(256) void k_gemm(const u16* __restrict__ A,
                                              const u16* __restrict__ W,
                                              const float* __restrict__ bias,
                                              const float* __restrict__ res,
                                              float* __restrict__ C,
                                              u16* __restrict__ Cb,
                                              int M, int N, int K, int flags) {
  __shared__ __attribute__((aligned(16))) u16 As[2][128][40];
  __shared__ __attribute__((aligned(16))) u16 Ws[2][128][40];
  const int tid  = threadIdx.x;
  const int lane = tid & 31;
  const int wid  = tid >> 5;
  const int m0 = blockIdx.y * 128;
  const int n0 = blockIdx.x * 128;
  const int mw = (wid >> 2) * 64;      // wave M offset (0/64)
  const int nw = (wid & 3) * 32;       // wave N offset (0..96)
  const int l16 = lane & 15;
  const int hi  = lane >> 4;

  // global staging: 256 threads cover 128 rows x 4 chunks (x2 slots)
  const int arow = tid >> 2;           // 0..63
  const int ac   = (tid & 3) * 8;      // elem offset of 16B chunk
  const u16* Ag = A + (size_t)(m0 + arow) * K + ac;
  const u16* Wg = W + (size_t)(n0 + arow) * K + ac;
  const size_t str64 = (size_t)64 * K;
  const int nk = K >> 5;

  f32x8 acc[4][2];
  const f32x8 fz = {0.f, 0.f, 0.f, 0.f, 0.f, 0.f, 0.f, 0.f};
#pragma unroll
  for (int i = 0; i < 4; i++) { acc[i][0] = fz; acc[i][1] = fz; }

  auto compute = [&](int buf) {
    Frag af[4], wf[2];
#pragma unroll
    for (int mi = 0; mi < 4; mi++) {
      const u16* p = &As[buf][mw + mi * 16 + l16][0];
      af[mi].q[0] = *(const v4u*)(p + hi * 16);       // K = hi*16 .. +7
      af[mi].q[1] = *(const v4u*)(p + hi * 16 + 8);   // K = hi*16+8 .. +15
    }
#pragma unroll
    for (int ni = 0; ni < 2; ni++) {
      const u16* p = &Ws[buf][nw + ni * 16 + l16][hi * 16];  // K = hi*16 .. +15
      wf[ni].q[0] = *(const v4u*)(p);
      wf[ni].q[1] = *(const v4u*)(p + 8);
    }
#pragma unroll
    for (int mi = 0; mi < 4; mi++)
#pragma unroll
      for (int ni = 0; ni < 2; ni++)
        acc[mi][ni] = wmma_bf16(af[mi].v, wf[ni].v, acc[mi][ni]);
  };

#if USE_ASYNC_LDS
  // CDNA5 path: GLOBAL_LOAD_ASYNC_TO_LDS_B128, tracked by ASYNCcnt.
  auto issue4 = [&](int step) {
    const int kk = step << 5;
    const int buf = step & 1;
    __builtin_amdgcn_global_load_async_to_lds_b128(
        (AS_GLOBAL v4i*)(Ag + kk),
        (AS_LDS v4i*)&As[buf][arow][ac], 0, 0);
    __builtin_amdgcn_global_load_async_to_lds_b128(
        (AS_GLOBAL v4i*)(Ag + str64 + kk),
        (AS_LDS v4i*)&As[buf][arow + 64][ac], 0, 0);
    __builtin_amdgcn_global_load_async_to_lds_b128(
        (AS_GLOBAL v4i*)(Wg + kk),
        (AS_LDS v4i*)&Ws[buf][arow][ac], 0, 0);
    __builtin_amdgcn_global_load_async_to_lds_b128(
        (AS_GLOBAL v4i*)(Wg + str64 + kk),
        (AS_LDS v4i*)&Ws[buf][arow + 64][ac], 0, 0);
  };
  issue4(0);
  for (int it = 0; it < nk; ++it) {
    __builtin_amdgcn_s_wait_asynccnt(0);  // own tile-it loads resident in LDS
    __syncthreads();                      // all waves resident + prev-buf reads retired
    if (it + 1 < nk) issue4(it + 1);      // overlap next tile with compute
    compute(it & 1);
  }
#else
  v4u ra0, ra1, rw0, rw1;
  ra0 = *(const v4u*)(Ag);
  ra1 = *(const v4u*)(Ag + str64);
  rw0 = *(const v4u*)(Wg);
  rw1 = *(const v4u*)(Wg + str64);
  for (int it = 0; it < nk; ++it) {
    const int buf = it & 1;
    *(v4u*)&As[buf][arow][ac]      = ra0;
    *(v4u*)&As[buf][arow + 64][ac] = ra1;
    *(v4u*)&Ws[buf][arow][ac]      = rw0;
    *(v4u*)&Ws[buf][arow + 64][ac] = rw1;
    if (it + 2 < nk) {                    // gfx1250 prefetch of tile it+2
      __builtin_prefetch(Ag + ((it + 2) << 5), 0, 0);
      __builtin_prefetch(Wg + ((it + 2) << 5), 0, 0);
    }
    if (it + 1 < nk) {
      const int kk = (it + 1) << 5;
      ra0 = *(const v4u*)(Ag + kk);
      ra1 = *(const v4u*)(Ag + str64 + kk);
      rw0 = *(const v4u*)(Wg + kk);
      rw1 = *(const v4u*)(Wg + str64 + kk);
    }
    __syncthreads();
    compute(buf);
  }
#endif

#pragma unroll
  for (int mi = 0; mi < 4; mi++) {
    const int mb = m0 + mw + mi * 16 + hi * 8;
#pragma unroll
    for (int ni = 0; ni < 2; ni++) {
      const int n = n0 + nw + ni * 16 + l16;
      const float bv = bias ? bias[n] : 0.f;
#pragma unroll
      for (int r = 0; r < 8; r++) {
        const size_t o = (size_t)(mb + r) * N + n;
        float t = acc[mi][ni][r] + bv;
        if (flags & GF_RELU) t = fmaxf(t, 0.f);
        if (flags & GF_ACC)  t += C[o];
        if (res)             t = (flags & GF_HALF) ? 0.5f * (t + res[o]) : t + res[o];
        if (Cb) Cb[o] = f2bf(t);
        if (C)  C[o]  = t;
      }
    }
  }
}

// ---------------- flash attention: one wave per (b,h,q-tile of 16) ----------------

__global__ __launch_bounds__(32) void k_attn(const u16* __restrict__ Qh,
                                             const u16* __restrict__ Kh,
                                             const u16* __restrict__ VhT,
                                             const unsigned char* __restrict__ kmask,
                                             u16* __restrict__ O) {
  __shared__ __attribute__((aligned(16))) u16 pt[16][40];
  const int lane = threadIdx.x & 31;
  const int l16 = lane & 15, hi = lane >> 4;
  int bid = blockIdx.x;
  const int q0 = (bid & 31) << 4; bid >>= 5;
  const int h = bid & 7;
  const int b = bid >> 3;
  const size_t bh = (size_t)(b * NHEAD + h);
  const u16* Kbase = Kh + bh * SEQL * HD;
  const u16* Vbase = VhT + bh * HD * SEQL;

  // Q tile fragments (A-layout), K-chunks d=0..31 and d=32..63
  Frag qa0, qa1;
  {
    const u16* p = Qh + (bh * SEQL + q0 + l16) * HD;
    qa0.q[0] = *(const v4u*)(p + hi * 16);
    qa0.q[1] = *(const v4u*)(p + 8 + hi * 16);
    qa1.q[0] = *(const v4u*)(p + 32 + hi * 16);
    qa1.q[1] = *(const v4u*)(p + 40 + hi * 16);
  }

  const f32x8 fz = {0.f, 0.f, 0.f, 0.f, 0.f, 0.f, 0.f, 0.f};
  f32x8 ao[4];
#pragma unroll
  for (int t = 0; t < 4; t++) ao[t] = fz;
  float rmax[8], rsum[8];
#pragma unroll
  for (int r = 0; r < 8; r++) { rmax[r] = -1e30f; rsum[r] = 0.f; }

  const int nkb = (q0 + 47) >> 5;     // ceil((q0+16)/32)
  for (int kb = 0; kb < nkb * 32; kb += 32) {
    // scores: two 16x16 tiles (keys kb..+15, kb+16..+31), contraction over d (2 x K=32)
    f32x8 sa = fz, sb = fz;
    Frag kf;
    const u16* kp;
    kp = Kbase + (size_t)(kb + l16) * HD + hi * 16;
    kf.q[0] = *(const v4u*)(kp); kf.q[1] = *(const v4u*)(kp + 8);
    sa = wmma_bf16(qa0.v, kf.v, sa);
    kp = Kbase + (size_t)(kb + l16) * HD + 32 + hi * 16;
    kf.q[0] = *(const v4u*)(kp); kf.q[1] = *(const v4u*)(kp + 8);
    sa = wmma_bf16(qa1.v, kf.v, sa);
    kp = Kbase + (size_t)(kb + 16 + l16) * HD + hi * 16;
    kf.q[0] = *(const v4u*)(kp); kf.q[1] = *(const v4u*)(kp + 8);
    sb = wmma_bf16(qa0.v, kf.v, sb);
    kp = Kbase + (size_t)(kb + 16 + l16) * HD + 32 + hi * 16;
    kf.q[0] = *(const v4u*)(kp); kf.q[1] = *(const v4u*)(kp + 8);
    sb = wmma_bf16(qa1.v, kf.v, sb);

    const int ka = kb + l16, kbk = kb + 16 + l16;
    const bool ma = kmask[b * SEQL + ka] != 0;
    const bool mb = kmask[b * SEQL + kbk] != 0;

    float alph[8];
#pragma unroll
    for (int r = 0; r < 8; r++) {
      const int q = q0 + hi * 8 + r;
      const float va = (ka > q || ma) ? -1e30f : sa[r];
      const float vb = (kbk > q || mb) ? -1e30f : sb[r];
      float tmax = half_max(fmaxf(va, vb));
      const float nm = fmaxf(rmax[r], tmax);
      alph[r] = __expf(rmax[r] - nm);
      const float pa = __expf(va - nm);
      const float pb = __expf(vb - nm);
      rsum[r] = rsum[r] * alph[r] + half_sum(pa + pb);
      rmax[r] = nm;
      pt[hi * 8 + r][l16]      = f2bf(pa);
      pt[hi * 8 + r][16 + l16] = f2bf(pb);
    }
#pragma unroll
    for (int t = 0; t < 4; t++)
#pragma unroll
      for (int r = 0; r < 8; r++) ao[t][r] *= alph[r];

    __syncthreads();
    Frag pf;
    {
      const u16* p = &pt[l16][0];
      pf.q[0] = *(const v4u*)(p + hi * 16);
      pf.q[1] = *(const v4u*)(p + 8 + hi * 16);
    }
    __syncthreads();

#pragma unroll
    for (int t = 0; t < 4; t++) {
      Frag vf;
      const u16* vp = Vbase + (size_t)(t * 16 + l16) * SEQL + kb + hi * 16;
      vf.q[0] = *(const v4u*)(vp); vf.q[1] = *(const v4u*)(vp + 8);
      ao[t] = wmma_bf16(pf.v, vf.v, ao[t]);
    }
  }

  float inv[8];
#pragma unroll
  for (int r = 0; r < 8; r++) inv[r] = 1.f / rsum[r];
#pragma unroll
  for (int t = 0; t < 4; t++)
#pragma unroll
    for (int r = 0; r < 8; r++) {
      const int s = q0 + hi * 8 + r;
      const int col = h * HD + t * 16 + l16;
      O[(size_t)(s * BATCH + b) * EMB + col] = f2bf(ao[t][r] * inv[r]);
    }
}

// ---------------------------------------------------------------------------

extern "C" void kernel_launch(void* const* d_in, const int* in_sizes, int n_in,
                              void* d_out, int out_size, void* d_ws, size_t ws_size,
                              hipStream_t stream) {
  const float* x_in        = (const float*)d_in[0];
  const float* hidden_prev = (const float*)d_in[1];
  const float* stack_prev  = (const float*)d_in[2];
  const unsigned char* k_mask = (const unsigned char*)d_in[3];
  const float* in_proj_w  = (const float*)d_in[4];
  const float* in_proj_b  = (const float*)d_in[5];
  const float* out_proj_w = (const float*)d_in[6];
  const float* out_proj_b = (const float*)d_in[7];
  const float* ln1_g = (const float*)d_in[8];
  const float* ln1_b = (const float*)d_in[9];
  const float* ln2_g = (const float*)d_in[10];
  const float* ln2_b = (const float*)d_in[11];
  const float* ff_w1 = (const float*)d_in[12];
  const float* ff_b1 = (const float*)d_in[13];
  const float* ff_w2 = (const float*)d_in[14];
  const float* ff_b2 = (const float*)d_in[15];
  const float* W_w = (const float*)d_in[16];
  const float* W_b = (const float*)d_in[17];
  const float* R_w = (const float*)d_in[18];
  const float* R_b = (const float*)d_in[19];
  const float* P_w = (const float*)d_in[20];
  const float* P_b = (const float*)d_in[21];
  const float* V_w = (const float*)d_in[22];
  const float* U_w = (const float*)d_in[23];
  const float* A_w = (const float*)d_in[24];
  const float* A_b = (const float*)d_in[25];
  const float* D_w = (const float*)d_in[26];
  const float* D_b = (const float*)d_in[27];

  float* out_x      = (float*)d_out;
  float* out_hidden = out_x + (size_t)NTOK * EMB;
  float* out_stack  = out_hidden + (size_t)NTOK * DHID;

  char* wsb = (char*)d_ws;
  size_t off = 0;
  auto alloc = [&](size_t bytes) -> void* {
    void* p = wsb + off;
    off = (off + bytes + 255) & ~(size_t)255;
    return p;
  };

  // bf16 weight copies
  u16* w_in  = (u16*)alloc((size_t)1536 * 512 * 2);
  u16* w_op  = (u16*)alloc((size_t)512 * 512 * 2);
  u16* w_Wb  = (u16*)alloc((size_t)512 * 512 * 2);
  u16* w_Rb  = (u16*)alloc((size_t)512 * 512 * 2);
  u16* w_Pb  = (u16*)alloc((size_t)512 * 128 * 2);
  u16* w_Vb  = (u16*)alloc((size_t)128 * 1024 * 2);
  u16* w_Ub  = (u16*)alloc((size_t)512 * 128 * 2);
  u16* w_Db  = (u16*)alloc((size_t)128 * 512 * 2);
  u16* w_f1  = (u16*)alloc((size_t)2048 * 512 * 2);
  u16* w_f2  = (u16*)alloc((size_t)512 * 2048 * 2);
  // activations
  u16*   xln   = (u16*)alloc((size_t)NTOK * EMB * 2);
  u16*   qkvb  = (u16*)alloc((size_t)NTOK * 1536 * 2);
  u16*   Qh    = (u16*)alloc((size_t)NTOK * EMB * 2);
  u16*   Kh    = (u16*)alloc((size_t)NTOK * EMB * 2);
  u16*   VhT   = (u16*)alloc((size_t)NTOK * EMB * 2);
  u16*   atto  = (u16*)alloc((size_t)NTOK * EMB * 2);
  float* x1    = (float*)alloc((size_t)NTOK * EMB * 4);
  u16*   x1u   = (u16*)alloc((size_t)NTOK * EMB * 2);
  u16*   hpv   = (u16*)alloc((size_t)NTOK * DHID * 2);
  u16*   stx   = (u16*)alloc((size_t)NTOK * SWID * 2);
  float* hpre  = (float*)alloc((size_t)NTOK * DHID * 4);
  u16*   hidb  = (u16*)alloc((size_t)NTOK * DHID * 2);
  u16*   vxb   = (u16*)alloc((size_t)NTOK * (EMB + DHID) * 2);
  float* vz    = (float*)alloc((size_t)NTOK * DSS * 4);
  u16*   pbf   = (u16*)alloc((size_t)NTOK * DSS * 2);
  float* x2    = (float*)alloc((size_t)NTOK * EMB * 4);
  float* dpre  = (float*)alloc((size_t)NTOK * SWID * 4);
  float* sinp  = (float*)alloc((size_t)NTOK * SWID * 4);
  float* ctrl  = (float*)alloc((size_t)NTOK * 3 * 4);
  u16*   h2b   = (u16*)alloc((size_t)NTOK * EMB * 2);
  u16*   ff1b  = (u16*)alloc((size_t)NTOK * DFF * 2);
  (void)ws_size; (void)in_sizes; (void)n_in; (void)out_size;

  auto cvt = [&](const float* s, u16* d, int n) {
    k_cvt_bf16<<<(n + 255) / 256, 256, 0, stream>>>(s, d, n);
  };
  auto gemm = [&](const u16* A, const u16* W, const float* bias, const float* res,
                  float* C, u16* Cb, int M, int N, int K, int flags) {
    dim3 g(N / 128, M / 128);
    k_gemm<<<g, 256, 0, stream>>>(A, W, bias, res, C, Cb, M, N, K, flags);
  };

  // weight conversions
  cvt(in_proj_w,  w_in, 1536 * 512);
  cvt(out_proj_w, w_op, 512 * 512);
  cvt(W_w, w_Wb, 512 * 512);
  cvt(R_w, w_Rb, 512 * 512);
  cvt(P_w, w_Pb, 512 * 128);
  cvt(V_w, w_Vb, 128 * 1024);
  cvt(U_w, w_Ub, 512 * 128);
  cvt(D_w, w_Db, 128 * 512);
  cvt(ff_w1, w_f1, 2048 * 512);
  cvt(ff_w2, w_f2, 512 * 2048);

  // --- attention block ---
  k_ln<<<NTOK / 8, 256, 0, stream>>>(x_in, ln1_g, ln1_b, xln, NTOK);
  gemm(xln, w_in, in_proj_b, nullptr, nullptr, qkvb, NTOK, 1536, 512, 0);
  k_qkv_split<<<(NTOK * EMB + 255) / 256, 256, 0, stream>>>(qkvb, Qh, Kh, VhT);
  k_attn<<<BATCH * NHEAD * (SEQL / 16), 32, 0, stream>>>(Qh, Kh, VhT, k_mask, atto);
  gemm(atto, w_op, out_proj_b, x_in, x1, nullptr, NTOK, 512, 512, 0); // x1 = x_in + mha

  // --- hidden update ---
  k_gather_u<<<(NTOK * EMB + 255) / 256, 256, 0, stream>>>(x1, x1u);
  cvt(hidden_prev, hpv, NTOK * DHID);
  k_stackx<<<(NTOK * SWID + 255) / 256, 256, 0, stream>>>(stack_prev, stx);
  gemm(x1u, w_Wb, W_b, nullptr, hpre, nullptr, NTOK, 512, 512, 0);
  gemm(hpv, w_Rb, R_b, nullptr, hpre, nullptr, NTOK, 512, 512, GF_ACC);
  gemm(stx, w_Pb, P_b, nullptr, hpre, nullptr, NTOK, 512, 128, GF_ACC);
  k_nonsat<<<(NTOK * DHID + 255) / 256, 256, 0, stream>>>(hpre, out_hidden, hidb, NTOK * DHID);

  // --- softmax bottleneck: x2 = 0.5*(x1 + softmax(vx@V^T)@U^T) (fused in U epilogue) ---
  k_vx_build<<<(NTOK * 1024 + 255) / 256, 256, 0, stream>>>(x1, out_hidden, vxb);
  gemm(vxb, w_Vb, nullptr, nullptr, vz, nullptr, NTOK, 128, 1024, 0);
  k_softmax128<<<NTOK / 8, 256, 0, stream>>>(vz, pbf, NTOK);
  gemm(pbf, w_Ub, nullptr, x1, x2, nullptr, NTOK, 512, 128, GF_HALF);

  // --- stack update ---
  gemm(hidb, w_Db, D_b, nullptr, dpre, nullptr, NTOK, 128, 512, 0);
  k_nonsat<<<(NTOK * SWID + 255) / 256, 256, 0, stream>>>(dpre, sinp, nullptr, NTOK * SWID);
  k_controls<<<NTOK / 8, 256, 0, stream>>>(out_hidden, A_w, A_b, ctrl, NTOK);
  k_stack_update<<<(NTOK * SDEP * 32 + 255) / 256, 256, 0, stream>>>(stack_prev, sinp, ctrl, out_stack);

  // --- FFN ---
  k_ln<<<NTOK / 8, 256, 0, stream>>>(x2, ln2_g, ln2_b, h2b, NTOK);
  gemm(h2b, w_f1, ff_b1, nullptr, nullptr, ff1b, NTOK, 2048, 512, GF_RELU);
  gemm(ff1b, w_f2, ff_b2, x2, out_x, nullptr, NTOK, 512, 2048, 0);
}
